// MPNN_68899865362891
// MI455X (gfx1250) — matrix operand order repaired and checked by reading, not compiled, not run on hardware
//
#include <hip/hip_runtime.h>

typedef __attribute__((ext_vector_type(16))) __bf16 v16bf;
typedef __attribute__((ext_vector_type(8)))  __bf16 bf16x8;
typedef __attribute__((ext_vector_type(4)))  __bf16 bf16x4;
typedef __attribute__((ext_vector_type(8)))  float  v8f;
typedef int v4i_g __attribute__((vector_size(16)));   // matches builtin's V4i param type

#define D 128
#define AS1 __attribute__((address_space(1)))
#define AS3 __attribute__((address_space(3)))

#if __has_builtin(__builtin_amdgcn_global_load_async_to_lds_b128)
#define HAVE_ASYNC_LDS 1
#else
#define HAVE_ASYNC_LDS 0
#endif

__device__ __forceinline__ void wait_dscnt0() {
#if __has_builtin(__builtin_amdgcn_s_wait_dscnt)
  __builtin_amdgcn_s_wait_dscnt(0);
#else
  asm volatile("s_wait_dscnt 0x0" ::: "memory");
#endif
}
__device__ __forceinline__ void wait_asynccnt0() {
#if __has_builtin(__builtin_amdgcn_s_wait_asynccnt)
  __builtin_amdgcn_s_wait_asynccnt(0);
#else
  asm volatile("s_wait_asynccnt 0x0" ::: "memory");
#endif
}

// ---- WMMA fragment loaders (bf16 16x16x32), layouts per CDNA5 ISA 7.12.2 ----
// A from an fp32 LDS tile (rows, stride in floats), converting to bf16:
//   lane L<16  : row L,    K in [kb..kb+7]    and [kb+16..kb+23]
//   lane L>=16 : row L-16, K in [kb+8..kb+15] and [kb+24..kb+31]
__device__ __forceinline__ v16bf loadA_f32(const float* base, int stride, int kb, int lane) {
  int row  = lane & 15;
  int koff = (lane < 16) ? 0 : 8;
  const float* p = base + row * stride + kb + koff;
  v16bf r;
  #pragma unroll
  for (int i = 0; i < 8; ++i) r[i] = (__bf16)p[i];
  #pragma unroll
  for (int i = 0; i < 8; ++i) r[8 + i] = (__bf16)p[16 + i];
  return r;
}
// A from a bf16 LDS tile (readout kernel)
__device__ __forceinline__ v16bf loadA(const __bf16* base, int stride, int kb, int lane) {
  int row  = lane & 15;
  int koff = (lane < 16) ? 0 : 8;
  const __bf16* p = base + row * stride + kb + koff;
  bf16x8 c0 = *(const bf16x8*)(p);
  bf16x8 c1 = *(const bf16x8*)(p + 16);
  return __builtin_shufflevector(c0, c1, 0,1,2,3,4,5,6,7,8,9,10,11,12,13,14,15);
}
// B (KxN tile with B[k][j] = W[j][k], W row-major bf16 in LDS):
//   lane n<16  : col j = jb+n,    K in [kb..kb+7]    and [kb+16..kb+23]
//   lane n>=16 : col j = jb+n-16, K in [kb+8..kb+15] and [kb+24..kb+31]
__device__ __forceinline__ v16bf loadB(const __bf16* base, int stride, int jb, int kb, int lane) {
  int j    = jb + (lane & 15);
  int koff = (lane < 16) ? 0 : 8;
  const __bf16* p = base + j * stride + kb + koff;
  bf16x8 c0 = *(const bf16x8*)(p);
  bf16x8 c1 = *(const bf16x8*)(p + 16);
  return __builtin_shufflevector(c0, c1, 0,1,2,3,4,5,6,7,8,9,10,11,12,13,14,15);
}

// ---- fp32 -> bf16 weight conversion (once per launch) ----
__global__ void cvt_weights_kernel(const float* __restrict__ Wm,
                                   const float* __restrict__ W1,
                                   const float* __restrict__ W2,
                                   __bf16* __restrict__ WmB,
                                   __bf16* __restrict__ W1B,
                                   __bf16* __restrict__ W2B,
                                   int nWm, int nW) {
  int i = blockIdx.x * blockDim.x + threadIdx.x;
  if (i < nWm) WmB[i] = (__bf16)Wm[i];
  if (i < nW)  { W1B[i] = (__bf16)W1[i]; W2B[i] = (__bf16)W2[i]; }
}

// ---- edge message kernel: one round. 4 waves/block, grid-stride 16-edge tiles ----
// msg = relu(concat(lat[from],lat[to]) @ Wm^T + bm) * scale; atomicAdd into latOut[to]
#define HS  264   // Wm LDS stride (bf16 elements), 256 + 8 pad
#define HSF 260   // h-tile LDS stride (fp32 elements), 256 + 4 pad
__global__ __launch_bounds__(128) void edge_msg_kernel(
    const float* __restrict__ latIn, float* __restrict__ latOut,
    const __bf16* __restrict__ WmB,  const float* __restrict__ bmR,
    const int* __restrict__ eFrom,   const int* __restrict__ eTo,
    const int* __restrict__ chPtr,   int E)
{
  __shared__ __bf16 sW[128 * HS];       // Wm[r] : [128][256] row-major, padded
  __shared__ float  sHf[4][16 * HSF];   // per-wave fp32 edge-feature tile [16][256]

  int tid = threadIdx.x;
  // stage Wm (one row per thread, 32x 16B chunks)
  {
    int r = tid;  // blockDim == 128 rows
    const bf16x8* src = (const bf16x8*)(WmB + r * 256);
    bf16x8* dst = (bf16x8*)(sW + r * HS);
    #pragma unroll
    for (int i = 0; i < 32; ++i) dst[i] = src[i];
  }
  __syncthreads();

  int wave = tid >> 5, lane = tid & 31;
  float scale = 1.0f / (float)(*chPtr - 1);
  int numTiles = (E + 15) >> 4;
  float* h = sHf[wave];
  int mbase = (lane < 16) ? 0 : 8;

  for (int tile = blockIdx.x * 4 + wave; tile < numTiles; tile += gridDim.x * 4) {
    int tileBase = tile * 16;
    int cnt = E - tileBase; if (cnt > 16) cnt = 16;

    int ef = 0, et = 0;
    if (lane < 16) {
      int li = (lane < cnt) ? lane : (cnt - 1);
      ef = eFrom[tileBase + li];
      et = eTo[tileBase + li];
    }

    // gather: 2 lanes per edge row; each lane moves 128 floats (512B) into LDS
    int row = lane >> 1, half = lane & 1;
    int idx = half ? __shfl(et, row) : __shfl(ef, row);
    const float* src = latIn + (size_t)idx * D;
    float* dst = h + row * HSF + half * D;

    wait_dscnt0();   // prior tile's LDS reads drained before overwriting h
#if HAVE_ASYNC_LDS
    {
      AS1 v4i_g* gp = (AS1 v4i_g*)src;
      AS3 v4i_g* lp = (AS3 v4i_g*)dst;
      #pragma unroll
      for (int i = 0; i < 32; ++i) {
        __builtin_amdgcn_global_load_async_to_lds_b128(gp, lp, 0, 0);
        gp += 1; lp += 1;
      }
    }
    wait_asynccnt0();
    asm volatile("" ::: "memory");
#else
    {
      const float4* s4 = (const float4*)src;
      float4* d4 = (float4*)dst;
      #pragma unroll
      for (int i = 0; i < 32; ++i) d4[i] = s4[i];
    }
#endif

    // build A fragments once per tile (fp32 LDS -> bf16 regs)
    v16bf Afr[8];
    #pragma unroll
    for (int kc = 0; kc < 8; ++kc) Afr[kc] = loadA_f32(h, HSF, kc * 32, lane);

    // 16 edges x 128 outputs: 8 col-tiles processed in pairs (two independent
    // ds_load->wmma chains so B feeds stay software-pipelined), K=256 -> 8 WMMA each
    for (int jt = 0; jt < 8; jt += 2) {
      v8f acc0 = {}, acc1 = {};
      #pragma unroll
      for (int kc = 0; kc < 8; ++kc) {
        v16bf B0 = loadB(sW, HS, jt * 16,      kc * 32, lane);
        v16bf B1 = loadB(sW, HS, jt * 16 + 16, kc * 32, lane);
        acc0 = __builtin_amdgcn_wmma_f32_16x16x32_bf16(false, Afr[kc], false, B0,
                                                       (short)0, acc0, false, false);
        acc1 = __builtin_amdgcn_wmma_f32_16x16x32_bf16(false, Afr[kc], false, B1,
                                                       (short)0, acc1, false, false);
      }
      #pragma unroll
      for (int half2 = 0; half2 < 2; ++half2) {
        v8f acc = half2 ? acc1 : acc0;
        int j = (jt + half2) * 16 + (lane & 15);
        float bias = bmR[j];
        #pragma unroll
        for (int g = 0; g < 8; ++g) {
          int m = mbase + g;
          float v = fmaxf(acc[g] + bias, 0.0f) * scale;
          int t = __shfl(et, m);
          if (m < cnt) atomicAdd(latOut + (size_t)t * D + j, v);
        }
      }
    }
  }
}

// ---- segment-sum scatter: y[view_id[n]] += lat[n] ----
__global__ void seg_sum_kernel(const float* __restrict__ lat,
                               const int* __restrict__ vid,
                               float* __restrict__ y, int N) {
  int t = blockIdx.x * blockDim.x + threadIdx.x;
  int n = t >> 5, c = (t & 31) * 4;
  if (n >= N) return;
  int b = vid[n];
  float4 v = *(const float4*)(lat + (size_t)n * D + c);
  float* dst = y + (size_t)b * D + c;
  atomicAdd(dst + 0, v.x); atomicAdd(dst + 1, v.y);
  atomicAdd(dst + 2, v.z); atomicAdd(dst + 3, v.w);
}

// ---- fused readout: out = relu(y@W1^T+b1)@W2^T + b2 ; 16 rows per wave ----
#define RS 136   // 128 + 8 pad (bf16)
__global__ __launch_bounds__(128) void readout_kernel(
    const float* __restrict__ y,
    const __bf16* __restrict__ W1B, const float* __restrict__ b1,
    const __bf16* __restrict__ W2B, const float* __restrict__ b2,
    float* __restrict__ out, int Bn)
{
  __shared__ __bf16 sW1[128 * RS];
  __shared__ __bf16 sW2[128 * RS];
  __shared__ __bf16 sY[4][16 * RS];
  __shared__ __bf16 sHl[4][16 * RS];

  int tid = threadIdx.x;
  {
    int r = tid;
    const bf16x8* s1 = (const bf16x8*)(W1B + r * 128);
    const bf16x8* s2 = (const bf16x8*)(W2B + r * 128);
    bf16x8* d1 = (bf16x8*)(sW1 + r * RS);
    bf16x8* d2 = (bf16x8*)(sW2 + r * RS);
    #pragma unroll
    for (int i = 0; i < 16; ++i) { d1[i] = s1[i]; d2[i] = s2[i]; }
  }
  __syncthreads();

  int wave = tid >> 5, lane = tid & 31;
  int rowBase = (blockIdx.x * 4 + wave) * 16;
  if (rowBase >= Bn) return;                  // wave-uniform
  int cnt = Bn - rowBase; if (cnt > 16) cnt = 16;

  // stage y tile -> bf16 LDS
  {
    int row = lane >> 1, half = lane & 1;
    int rc = (row < cnt) ? row : (cnt - 1);
    const float4* src = (const float4*)(y + (size_t)(rowBase + rc) * D + half * 64);
    __bf16* dst = sY[wave] + row * RS + half * 64;
    #pragma unroll
    for (int i = 0; i < 16; ++i) {
      float4 v = src[i];
      bf16x4 p = { (__bf16)v.x, (__bf16)v.y, (__bf16)v.z, (__bf16)v.w };
      *(bf16x4*)(dst + i * 4) = p;
    }
  }

  __bf16* hT = sHl[wave];
  // layer 1: h = relu(y@W1^T + b1) -> bf16 LDS (A-layout source for layer 2)
  for (int jt = 0; jt < 8; ++jt) {
    v8f acc = {};
    #pragma unroll
    for (int kc = 0; kc < 4; ++kc) {
      v16bf A  = loadA(sY[wave], RS, kc * 32, lane);
      v16bf Bv = loadB(sW1,      RS, jt * 16, kc * 32, lane);
      acc = __builtin_amdgcn_wmma_f32_16x16x32_bf16(false, A, false, Bv,
                                                    (short)0, acc, false, false);
    }
    int j = jt * 16 + (lane & 15);
    float bias = b1[j];
    int mbase = (lane < 16) ? 0 : 8;
    #pragma unroll
    for (int g = 0; g < 8; ++g)
      hT[(mbase + g) * RS + j] = (__bf16)fmaxf(acc[g] + bias, 0.0f);
  }
  // layer 2: out = h@W2^T + b2 (fp32 stores)
  for (int jt = 0; jt < 8; ++jt) {
    v8f acc = {};
    #pragma unroll
    for (int kc = 0; kc < 4; ++kc) {
      v16bf A  = loadA(hT,  RS, kc * 32, lane);
      v16bf Bv = loadB(sW2, RS, jt * 16, kc * 32, lane);
      acc = __builtin_amdgcn_wmma_f32_16x16x32_bf16(false, A, false, Bv,
                                                    (short)0, acc, false, false);
    }
    int j = jt * 16 + (lane & 15);
    float bias = b2[j];
    int mbase = (lane < 16) ? 0 : 8;
    #pragma unroll
    for (int g = 0; g < 8; ++g) {
      int m = mbase + g;
      if (m < cnt) out[(size_t)(rowBase + m) * D + j] = acc[g] + bias;
    }
  }
}

extern "C" void kernel_launch(void* const* d_in, const int* in_sizes, int n_in,
                              void* d_out, int out_size, void* d_ws, size_t ws_size,
                              hipStream_t stream) {
  const float* latents = (const float*)d_in[0];
  const float* Wm  = (const float*)d_in[1];
  const float* bm  = (const float*)d_in[2];
  const float* W1  = (const float*)d_in[3];
  const float* b1  = (const float*)d_in[4];
  const float* W2  = (const float*)d_in[5];
  const float* b2  = (const float*)d_in[6];
  const int* eFrom = (const int*)d_in[7];
  const int* eTo   = (const int*)d_in[8];
  const int* vid   = (const int*)d_in[9];
  const int* chP   = (const int*)d_in[10];

  int N  = in_sizes[0] / D;
  int E  = in_sizes[7];
  int R  = in_sizes[2] / D;
  int Bn = out_size / D;
  int nWm = in_sizes[1];     // R * D * 2D
  int nW  = in_sizes[3];     // D * D

  size_t latBytes = (size_t)N * D * sizeof(float);
  size_t yBytes   = (size_t)Bn * D * sizeof(float);
  char* ws = (char*)d_ws;
  float* buf0 = (float*)ws;
  float* buf1 = (float*)(ws + latBytes);
  float* yBuf = (float*)(ws + 2 * latBytes);
  __bf16* WmB = (__bf16*)(ws + 2 * latBytes + yBytes);
  __bf16* W1B = WmB + nWm;
  __bf16* W2B = W1B + nW;

  cvt_weights_kernel<<<(nWm + 255) / 256, 256, 0, stream>>>(Wm, W1, W2, WmB, W1B, W2B, nWm, nW);

  (void)hipMemcpyAsync(buf0, latents, latBytes, hipMemcpyDeviceToDevice, stream);
  float* inB = buf0; float* outB = buf1;
  int numTiles = (E + 15) / 16;
  int edgeBlocks = (numTiles + 4 * 8 - 1) / (4 * 8);   // 8 tiles per wave
  for (int r = 0; r < R; ++r) {
    (void)hipMemcpyAsync(outB, inB, latBytes, hipMemcpyDeviceToDevice, stream);
    edge_msg_kernel<<<edgeBlocks, 128, 0, stream>>>(inB, outB,
        WmB + (size_t)r * D * 2 * D, bm + r * D, eFrom, eTo, chP, E);
    float* t = inB; inB = outB; outB = t;
  }

  (void)hipMemsetAsync(yBuf, 0, yBytes, stream);
  seg_sum_kernel<<<(N * 32 + 255) / 256, 256, 0, stream>>>(inB, vid, yBuf, N);
  readout_kernel<<<(Bn + 63) / 64, 128, 0, stream>>>(yBuf, W1B, b1, W2B, b2,
                                                     (float*)d_out, Bn);
}
